// GraphClusteringLoss_43224550868147
// MI455X (gfx1250) — matrix operand order repaired and checked by reading, not compiled, not run on hardware
//
#include <hip/hip_runtime.h>

typedef __attribute__((ext_vector_type(16))) _Float16     v16h;
typedef __attribute__((ext_vector_type(8)))  float        v8f;
typedef __attribute__((ext_vector_type(4)))  float        f32x4;
typedef __attribute__((ext_vector_type(4)))  unsigned int u32x4;

// clang's async-LDS builtin wants GCC-style int4 vectors with explicit AS
typedef int v4i_b __attribute__((vector_size(16)));
typedef __attribute__((address_space(1))) v4i_b* as1_v4i;
typedef __attribute__((address_space(3))) v4i_b* as3_v4i;

static constexpr int kN = 8192;   // graph nodes
static constexpr int kC = 64;     // clusters
static constexpr int kLDB = 136;  // padded LDS row stride (halves): 68 dwords -> bank stride 4

// ---------------------------------------------------------------------------
// CDNA5 async global->LDS copy (16 B per lane), ASYNCcnt-tracked.
// Builtin signature (from hipcc diagnostic): (v4i AS1*, v4i AS3*, imm, imm).
// LDS byte offset of a generic shared pointer is its low 32 bits (aperture rule).
// ---------------------------------------------------------------------------
__device__ __forceinline__ void async_copy16(const void* g, void* l) {
#if __has_builtin(__builtin_amdgcn_global_load_async_to_lds_b128)
    __builtin_amdgcn_global_load_async_to_lds_b128(
        (as1_v4i)(unsigned long long)g,
        (as3_v4i)(unsigned int)(unsigned long long)l,
        0, 0);
#else
    unsigned lds_off = (unsigned)(unsigned long long)l;
    asm volatile("global_load_async_to_lds_b128 %0, %1, off"
                 :: "v"(lds_off), "v"((unsigned long long)g)
                 : "memory");
#endif
}

__device__ __forceinline__ void async_wait0() {
#if __has_builtin(__builtin_amdgcn_s_wait_asynccnt)
    __builtin_amdgcn_s_wait_asynccnt(0);
#else
    asm volatile("s_wait_asynccnt 0x0" ::: "memory");
#endif
}

// ---------------------------------------------------------------------------
// Kernel 0: zero M (64x64 fp32) and nwc (64 fp32)
// ---------------------------------------------------------------------------
__global__ void k_init(float* __restrict__ M, float* __restrict__ nwc) {
    int t = threadIdx.x;
    for (int i = t; i < kC * kC; i += 256) M[i] = 0.0f;
    if (t < kC) nwc[t] = 0.0f;
}

// ---------------------------------------------------------------------------
// Kernel 1: row softmax of Y [N,64]; store Ys TRANSPOSED as f16 (YsT[c][n]),
// accumulate column sums nwc. One wave (32 lanes) per row, 2 cols per lane.
// ---------------------------------------------------------------------------
__global__ void k_softmax(const float* __restrict__ Y,
                          _Float16* __restrict__ YsT,
                          float* __restrict__ nwc) {
    __shared__ float s_nwc[kC];
    int tid  = threadIdx.x;
    int lane = tid & 31;
    int w    = tid >> 5;
    int row  = blockIdx.x * 8 + w;

    if (tid < kC) s_nwc[tid] = 0.0f;
    __syncthreads();

    float y0 = Y[(size_t)row * kC + lane];
    float y1 = Y[(size_t)row * kC + lane + 32];
    float m = fmaxf(y0, y1);
    #pragma unroll
    for (int o = 16; o > 0; o >>= 1) m = fmaxf(m, __shfl_xor(m, o, 32));
    float e0 = expf(y0 - m), e1 = expf(y1 - m);
    float s = e0 + e1;
    #pragma unroll
    for (int o = 16; o > 0; o >>= 1) s += __shfl_xor(s, o, 32);
    float inv = 1.0f / s;
    float ys0 = e0 * inv, ys1 = e1 * inv;

    YsT[(size_t)lane        * kN + row] = (_Float16)ys0;
    YsT[(size_t)(lane + 32) * kN + row] = (_Float16)ys1;

    atomicAdd(&s_nwc[lane],      ys0);
    atomicAdd(&s_nwc[lane + 32], ys1);
    __syncthreads();
    if (tid < kC) atomicAdd(&nwc[tid], s_nwc[tid]);
}

// ---------------------------------------------------------------------------
// Kernel 2 (hot): T = A @ Ys, A fp32 split into f16 hi+lo (2 WMMAs/tile).
// 128 blocks x 128 threads (4 waves); 16 rows per wave.
// Ys^T K-chunks double-buffered in LDS via async global->LDS copies;
// A stream software-pipelined through a rotating register set.
// ---------------------------------------------------------------------------
__global__ void __launch_bounds__(128)
k_AYs(const float* __restrict__ A, const _Float16* __restrict__ YsT,
      _Float16* __restrict__ TTh, _Float16* __restrict__ TTl) {
    __shared__ _Float16 sB[2][kC * kLDB];

    int tid  = threadIdx.x;
    int lane = tid & 31;
    int w    = tid >> 5;
    int half = lane >> 4;
    int l16  = lane & 15;

    int r0 = blockIdx.x * 64 + w * 16;          // 16-row tile for this wave
    const float* Arow = A + (size_t)(r0 + l16) * kN;

    v8f acc[4] = {};                            // 16x64 fp32 accumulators

    // staging assignment: thread -> (row sn, half-row hq), 64 halves = 128 B
    int sn = tid >> 1, hq = tid & 1;
    const _Float16* gsrc = YsT + (size_t)sn * kN + 64 * hq;
    _Float16* lrow0 = &sB[0][sn * kLDB + 64 * hq];
    _Float16* lrow1 = &sB[1][sn * kLDB + 64 * hq];

    // prologue: stage chunk 0 into buffer 0
    #pragma unroll
    for (int i = 0; i < 8; ++i) async_copy16(gsrc + 8 * i, lrow0 + 8 * i);
    async_wait0();
    __syncthreads();

    // prologue: preload A(k0 = 0)
    f32x4 a0 = *(const f32x4*)(Arow + 8 * half);
    f32x4 a1 = *(const f32x4*)(Arow + 8 * half + 4);
    f32x4 a2 = *(const f32x4*)(Arow + 16 + 8 * half);
    f32x4 a3 = *(const f32x4*)(Arow + 16 + 8 * half + 4);

    for (int kc = 0; kc < kN; kc += 128) {
        int cur = (kc >> 7) & 1;
        // issue async stage of next chunk into the other buffer (no wait yet)
        if (kc + 128 < kN) {
            const _Float16* gs = gsrc + kc + 128;
            _Float16* lr = cur ? lrow0 : lrow1;
            #pragma unroll
            for (int i = 0; i < 8; ++i) async_copy16(gs + 8 * i, lr + 8 * i);
            __builtin_prefetch(Arow + kc + 256, 0, 1);
        }
        const _Float16* sbuf = &sB[cur][0];

        #pragma unroll
        for (int s = 0; s < 4; ++s) {
            // convert current A registers: fp32 -> f16 hi + lo split
            float af[16];
            #pragma unroll
            for (int i = 0; i < 4; ++i) {
                af[i] = a0[i]; af[4 + i] = a1[i]; af[8 + i] = a2[i]; af[12 + i] = a3[i];
            }
            v16h ah, al;
            #pragma unroll
            for (int i = 0; i < 16; ++i) {
                _Float16 h = (_Float16)af[i];
                ah[i] = h;
                al[i] = (_Float16)(af[i] - (float)h);
            }
            // software pipeline: preload A for next K-step before the WMMA chain
            int nk = (s == 3) ? (kc + 128) : (kc + 32 * (s + 1));
            if (nk < kN) {
                a0 = *(const f32x4*)(Arow + nk + 8 * half);
                a1 = *(const f32x4*)(Arow + nk + 8 * half + 4);
                a2 = *(const f32x4*)(Arow + nk + 16 + 8 * half);
                a3 = *(const f32x4*)(Arow + nk + 16 + 8 * half + 4);
            }
            // hoist all four B tiles so one dscnt wait covers the WMMA chain
            union { u32x4 u[2]; v16h h; } b[4];
            #pragma unroll
            for (int g = 0; g < 4; ++g) {
                const u32x4* bp =
                    (const u32x4*)&sbuf[(g * 16 + l16) * kLDB + 32 * s + 8 * half];
                b[g].u[0] = bp[0]; b[g].u[1] = bp[2];
            }
            #pragma unroll
            for (int g = 0; g < 4; ++g) {
                acc[g] = __builtin_amdgcn_wmma_f32_16x16x32_f16(
                    false, ah, false, b[g].h, (short)0, acc[g], false, false);
                acc[g] = __builtin_amdgcn_wmma_f32_16x16x32_f16(
                    false, al, false, b[g].h, (short)0, acc[g], false, false);
            }
        }
        // next chunk's async stage must have landed before we flip buffers
        async_wait0();
        __syncthreads();
    }

    // write T tile transposed, split f16 hi/lo: TT[c][r] = T[r][c]
    #pragma unroll
    for (int g = 0; g < 4; ++g) {
        int col = g * 16 + l16;
        #pragma unroll
        for (int v = 0; v < 8; ++v) {
            int row = r0 + v + 8 * half;
            float val = acc[g][v];
            _Float16 th = (_Float16)val;
            _Float16 tl = (_Float16)(val - (float)th);
            TTh[(size_t)col * kN + row] = th;
            TTl[(size_t)col * kN + row] = tl;
        }
    }
}

// ---------------------------------------------------------------------------
// Kernel 3: M = Ys^T @ T (64x64, K=8192), split-K over 64 blocks of K=128.
// 8 waves x 2 output tiles each; M += Ys^T*(Th + Tl) via 2 WMMAs/tile/step.
// ---------------------------------------------------------------------------
__global__ void __launch_bounds__(256)
k_M(const _Float16* __restrict__ YsT, const _Float16* __restrict__ TTh,
    const _Float16* __restrict__ TTl, float* __restrict__ M) {
    int tid  = threadIdx.x;
    int lane = tid & 31;
    int w    = tid >> 5;
    int half = lane >> 4;
    int l16  = lane & 15;

    int kbase = blockIdx.x * 128;
    int gi  = w >> 1;                 // output row-tile
    int gj0 = 2 * (w & 1);            // first of two output col-tiles

    const _Float16* Ai  = YsT + (size_t)(gi * 16 + l16) * kN;
    const _Float16* Bh0 = TTh + (size_t)(gj0 * 16 + l16) * kN;
    const _Float16* Bh1 = TTh + (size_t)((gj0 + 1) * 16 + l16) * kN;
    const _Float16* Bl0 = TTl + (size_t)(gj0 * 16 + l16) * kN;
    const _Float16* Bl1 = TTl + (size_t)((gj0 + 1) * 16 + l16) * kN;

    v8f acc0 = {}, acc1 = {};
    #pragma unroll
    for (int s = 0; s < 4; ++s) {
        int k0 = kbase + 32 * s;
        union { u32x4 u[2]; v16h h; } a, b0h, b1h, b0l, b1l;
        a.u[0] = *(const u32x4*)(Ai + k0 + 16 * half);
        a.u[1] = *(const u32x4*)(Ai + k0 + 16 * half + 8);
        const u32x4* p;
        p = (const u32x4*)(Bh0 + k0 + 16 * half); b0h.u[0] = p[0]; b0h.u[1] = p[1];
        p = (const u32x4*)(Bh1 + k0 + 16 * half); b1h.u[0] = p[0]; b1h.u[1] = p[1];
        p = (const u32x4*)(Bl0 + k0 + 16 * half); b0l.u[0] = p[0]; b0l.u[1] = p[1];
        p = (const u32x4*)(Bl1 + k0 + 16 * half); b1l.u[0] = p[0]; b1l.u[1] = p[1];
        acc0 = __builtin_amdgcn_wmma_f32_16x16x32_f16(false, a.h, false, b0h.h, (short)0, acc0, false, false);
        acc0 = __builtin_amdgcn_wmma_f32_16x16x32_f16(false, a.h, false, b0l.h, (short)0, acc0, false, false);
        acc1 = __builtin_amdgcn_wmma_f32_16x16x32_f16(false, a.h, false, b1h.h, (short)0, acc1, false, false);
        acc1 = __builtin_amdgcn_wmma_f32_16x16x32_f16(false, a.h, false, b1l.h, (short)0, acc1, false, false);
    }
    #pragma unroll
    for (int v = 0; v < 8; ++v) {
        int i = gi * 16 + v + 8 * half;
        atomicAdd(&M[i * kC + gj0 * 16 + l16],       acc0[v]);
        atomicAdd(&M[i * kC + (gj0 + 1) * 16 + l16], acc1[v]);
    }
}

// ---------------------------------------------------------------------------
// Kernel 4: scalar epilogue. total_edges == sum(M) since softmax rows sum to 1
// (A.sum() == 1^T Ys^T A Ys 1). cohesion from diag(M)/max_ewc, coupling from
// strict upper triangle of M.
// ---------------------------------------------------------------------------
__global__ void k_final(const float* __restrict__ M, const float* __restrict__ nwc,
                        float* __restrict__ out) {
    __shared__ float s_tot[256], s_up[256], s_coh[kC];
    int t = threadIdx.x;
    float tot = 0.0f, up = 0.0f;
    for (int idx = t; idx < kC * kC; idx += 256) {
        float v = M[idx];
        int i = idx >> 6, j = idx & 63;
        tot += v;
        if (j > i) up += v;
    }
    s_tot[t] = tot; s_up[t] = up;
    if (t < kC) {
        float d  = M[t * kC + t];
        float nw = nwc[t];
        s_coh[t] = d / ((nw * (nw - 1.0f) + 1e-8f) * 0.5f);
    }
    __syncthreads();
    if (t == 0) {
        float T = 0.0f, U = 0.0f, Coh = 0.0f;
        for (int i = 0; i < 256; ++i) { T += s_tot[i]; U += s_up[i]; }
        for (int i = 0; i < kC; ++i) Coh += s_coh[i];
        out[0] = -Coh + U / (T + 1e-9f);
    }
}

// ---------------------------------------------------------------------------
extern "C" void kernel_launch(void* const* d_in, const int* in_sizes, int n_in,
                              void* d_out, int out_size, void* d_ws, size_t ws_size,
                              hipStream_t stream) {
    const float* A = (const float*)d_in[0];   // [8192, 8192] fp32
    const float* Y = (const float*)d_in[1];   // [8192, 64]   fp32
    float* out = (float*)d_out;               // scalar fp32

    char* ws = (char*)d_ws;
    _Float16* YsT = (_Float16*)(ws);                          // 64*8192 f16 = 1 MiB
    _Float16* TTh = (_Float16*)(ws + (1u << 20));             // 1 MiB
    _Float16* TTl = (_Float16*)(ws + (2u << 20));             // 1 MiB
    float*    M   = (float*)(ws + (3u << 20));                // 64*64 fp32
    float*    nwc = (float*)(ws + (3u << 20) + kC * kC * 4);  // 64 fp32

    hipLaunchKernelGGL(k_init,    dim3(1),    dim3(256), 0, stream, M, nwc);
    hipLaunchKernelGGL(k_softmax, dim3(kN/8), dim3(256), 0, stream, Y, YsT, nwc);
    hipLaunchKernelGGL(k_AYs,     dim3(kN/64),dim3(128), 0, stream, A, YsT, TTh, TTl);
    hipLaunchKernelGGL(k_M,       dim3(64),   dim3(256), 0, stream, YsT, TTh, TTl, M);
    hipLaunchKernelGGL(k_final,   dim3(1),    dim3(256), 0, stream, M, nwc, out);
}
